// FFTAttention_18348100288770
// MI455X (gfx1250) — matrix operand order repaired and checked by reading, not compiled, not run on hardware
//
// FFTAttention for MI455X (gfx1250, wave32, WMMA bf16 16x16x32, f32 accum).
//
// Pipeline (workspace holds bf16 intermediates; needs ~151 MB of d_ws):
//   [1] V16 = bf16( X @ Wv^T )                       (Wv = qkv_w rows [2C,3C))
//   [2] per (b,head,h):  P[v,d] = C[v,w]·V[w,d],  Q[v,d] = S[v,w]·V[w,d]
//   [3] per (b,head,v):  Y[u,d] = C[u,h]·P[h,d] + (-S)[u,h]·Q[h,d]   (= Re fft2)
//   [4] out = f32( Y16 @ proj_w^T + proj_b )
//
// Roofline: ~90 GFLOP vs ~600 MB HBM traffic -> bf16 WMMA keeps compute
// (~20-40us) balanced against memory (~26us @ 23.3 TB/s).

#include <hip/hip_runtime.h>
#include <hip/hip_bf16.h>

typedef __attribute__((ext_vector_type(16))) __bf16 v16bf;
typedef __attribute__((ext_vector_type(8)))  __bf16 v8bf;
typedef __attribute__((ext_vector_type(4)))  __bf16 v4bf;
typedef __attribute__((ext_vector_type(8)))  float  v8f;

#define SHUF16(lo, hi) \
  __builtin_shufflevector(lo, hi, 0,1,2,3,4,5,6,7,8,9,10,11,12,13,14,15)

// ---------------------------------------------------------------------------
// GEMM1: Out16[m,n] = bf16( sum_k A[m,k] * W[n,k] ),  A f32, W f32.
// Tile: 128(M) x 64(N) x 32(K); 256 threads = 8 waves, each wave 16x64.
// ---------------------------------------------------------------------------
__global__ __launch_bounds__(256) void gemm_f32in_bf16out(
    const float* __restrict__ A, const float* __restrict__ W,
    __bf16* __restrict__ Out, int M, int K, int Nout)
{
  __shared__ __bf16 Xs[128][40];   // 32 + 8 pad (row = 80B, 16B aligned)
  __shared__ __bf16 Ws[64][40];

  const int tid = threadIdx.x;
  const int m0 = blockIdx.x * 128;
  const int n0 = blockIdx.y * 64;
  const int wv = tid >> 5, lane = tid & 31, hi = lane >> 4, l16 = lane & 15;

  v8f zero = {};
  v8f acc[4] = { zero, zero, zero, zero };

  for (int k0 = 0; k0 < K; k0 += 32) {
    // Stage A tile 128x32 f32 -> bf16 (coalesced float4 loads)
    #pragma unroll
    for (int i = 0; i < 4; ++i) {
      int idx = i * 256 + tid;
      int row = idx >> 3;
      int c4  = (idx & 7) << 2;
      const float* p = A + (size_t)(m0 + row) * K + k0 + c4;
      float4 fv = *(const float4*)p;
      if (k0 + 32 < K) __builtin_prefetch(p + 32, 0, 1);   // global_prefetch_b8
      v4bf t = { (__bf16)fv.x, (__bf16)fv.y, (__bf16)fv.z, (__bf16)fv.w };
      *(v4bf*)&Xs[row][c4] = t;
    }
    // Stage W tile 64x32 f32 -> bf16
    #pragma unroll
    for (int i = 0; i < 2; ++i) {
      int idx = i * 256 + tid;
      int row = idx >> 3;
      int c4  = (idx & 7) << 2;
      float4 fv = *(const float4*)(W + (size_t)(n0 + row) * K + k0 + c4);
      v4bf t = { (__bf16)fv.x, (__bf16)fv.y, (__bf16)fv.z, (__bf16)fv.w };
      *(v4bf*)&Ws[row][c4] = t;
    }
    __syncthreads();

    // A fragment: lane m=l16, halves select K {0..7|8..15} and {16..23|24..31}
    const __bf16* arow = &Xs[wv * 16 + l16][0];
    v8bf alo = *(const v8bf*)(arow + hi * 8);
    v8bf ahi = *(const v8bf*)(arow + 16 + hi * 8);
    v16bf af = SHUF16(alo, ahi);
    #pragma unroll
    for (int j = 0; j < 4; ++j) {
      // B fragment: lane n=l16, halves select K {0..7|8..15} and {16..23|24..31}
      const __bf16* brow = &Ws[j * 16 + l16][0];
      v8bf blo = *(const v8bf*)(brow + hi * 8);
      v8bf bhi = *(const v8bf*)(brow + 16 + hi * 8);
      v16bf bfr = SHUF16(blo, bhi);
      acc[j] = __builtin_amdgcn_wmma_f32_16x16x32_bf16(
          false, af, false, bfr, (short)0, acc[j], false, false);
    }
    __syncthreads();
  }

  // C/D layout: element r of acc -> m = r + 8*hi, n = l16
  #pragma unroll
  for (int j = 0; j < 4; ++j) {
    int n = n0 + j * 16 + l16;
    #pragma unroll
    for (int r = 0; r < 8; ++r) {
      int m = m0 + wv * 16 + hi * 8 + r;
      Out[(size_t)m * Nout + n] = (__bf16)acc[j][r];
    }
  }
}

// ---------------------------------------------------------------------------
// GEMM2: Out[m,n] = sum_k A16[m,k] * W[n,k] + bias[n],  A bf16, Out f32.
// ---------------------------------------------------------------------------
__global__ __launch_bounds__(256) void gemm_bf16in_f32out(
    const __bf16* __restrict__ A, const float* __restrict__ W,
    const float* __restrict__ bias, float* __restrict__ Out,
    int M, int K, int Nout)
{
  __shared__ __bf16 Xs[128][40];
  __shared__ __bf16 Ws[64][40];

  const int tid = threadIdx.x;
  const int m0 = blockIdx.x * 128;
  const int n0 = blockIdx.y * 64;
  const int wv = tid >> 5, lane = tid & 31, hi = lane >> 4, l16 = lane & 15;

  v8f zero = {};
  v8f acc[4] = { zero, zero, zero, zero };

  for (int k0 = 0; k0 < K; k0 += 32) {
    // Stage A tile 128x32 bf16 (16B copies, already bf16)
    #pragma unroll
    for (int i = 0; i < 2; ++i) {
      int idx = i * 256 + tid;
      int row = idx >> 2;
      int c8  = (idx & 3) << 3;
      const __bf16* p = A + (size_t)(m0 + row) * K + k0 + c8;
      if (k0 + 32 < K) __builtin_prefetch(p + 32, 0, 1);
      *(v8bf*)&Xs[row][c8] = *(const v8bf*)p;
    }
    #pragma unroll
    for (int i = 0; i < 2; ++i) {
      int idx = i * 256 + tid;
      int row = idx >> 3;
      int c4  = (idx & 7) << 2;
      float4 fv = *(const float4*)(W + (size_t)(n0 + row) * K + k0 + c4);
      v4bf t = { (__bf16)fv.x, (__bf16)fv.y, (__bf16)fv.z, (__bf16)fv.w };
      *(v4bf*)&Ws[row][c4] = t;
    }
    __syncthreads();

    const __bf16* arow = &Xs[wv * 16 + l16][0];
    v8bf alo = *(const v8bf*)(arow + hi * 8);
    v8bf ahi = *(const v8bf*)(arow + 16 + hi * 8);
    v16bf af = SHUF16(alo, ahi);
    #pragma unroll
    for (int j = 0; j < 4; ++j) {
      const __bf16* brow = &Ws[j * 16 + l16][0];
      v8bf blo = *(const v8bf*)(brow + hi * 8);
      v8bf bhi = *(const v8bf*)(brow + 16 + hi * 8);
      v16bf bfr = SHUF16(blo, bhi);
      acc[j] = __builtin_amdgcn_wmma_f32_16x16x32_bf16(
          false, af, false, bfr, (short)0, acc[j], false, false);
    }
    __syncthreads();
  }

  #pragma unroll
  for (int j = 0; j < 4; ++j) {
    int n = n0 + j * 16 + l16;
    float bv = bias[n];
    #pragma unroll
    for (int r = 0; r < 8; ++r) {
      int m = m0 + wv * 16 + hi * 8 + r;
      Out[(size_t)m * Nout + n] = acc[j][r] + bv;
    }
  }
}

// ---------------------------------------------------------------------------
// FFT stage 1: one block per (b*12+head, h). P/Q layout: [bh][v][h][d].
// Waves 0-3: P = C @ Vt, waves 4-7: Q = S @ Vt  (each wave: 16(v) x 64(d)).
// ---------------------------------------------------------------------------
__global__ __launch_bounds__(256) void fft64_stage1(
    const __bf16* __restrict__ V, __bf16* __restrict__ P, __bf16* __restrict__ Q)
{
  __shared__ __bf16 Cs[64][72];
  __shared__ __bf16 Ss[64][72];
  __shared__ __bf16 Vt[64][72];   // [d][w] (transposed so B K-runs are contiguous)

  const int tid = threadIdx.x;
  const int bid = blockIdx.x;
  const int h  = bid & 63;
  const int bh = bid >> 6;               // b*12 + head
  const int b = bh / 12, head = bh % 12;

  // Twiddles: C[u][w]=cos(2pi*u*w/64), S=sin; phase reduced mod 64.
  #pragma unroll
  for (int i = 0; i < 16; ++i) {
    int e = i * 256 + tid;
    int u = e >> 6, w = e & 63;
    float ang = (float)((u * w) & 63) * 0.0981747704246810387f;  // 2pi/64
    float sv, cv;
    __sincosf(ang, &sv, &cv);
    Cs[u][w] = (__bf16)cv;
    Ss[u][w] = (__bf16)sv;
  }

  // Load V[b, h*64+w, head*64+d] -> Vt[d][w]
  {
    const __bf16* src = V + ((size_t)(b * 4096 + h * 64) * 768 + head * 64);
    int w  = tid >> 2;
    int d0 = (tid & 3) << 4;
    v8bf v0 = *(const v8bf*)(src + (size_t)w * 768 + d0);
    v8bf v1 = *(const v8bf*)(src + (size_t)w * 768 + d0 + 8);
    #pragma unroll
    for (int i = 0; i < 8; ++i) {
      Vt[d0 + i][w]     = v0[i];
      Vt[d0 + 8 + i][w] = v1[i];
    }
  }
  __syncthreads();

  const int wv = tid >> 5, lane = tid & 31, hi = lane >> 4, l16 = lane & 15;
  const int ublk = wv & 3;
  const bool doSin = wv >= 4;

  v8f zero = {};
  v8f acc[4] = { zero, zero, zero, zero };

  #pragma unroll
  for (int ks = 0; ks < 2; ++ks) {
    int kb = ks * 32;
    const __bf16* arow = (doSin ? &Ss[0][0] : &Cs[0][0]) + (ublk * 16 + l16) * 72;
    v8bf alo = *(const v8bf*)(arow + kb + hi * 8);
    v8bf ahi = *(const v8bf*)(arow + kb + 16 + hi * 8);
    v16bf af = SHUF16(alo, ahi);
    #pragma unroll
    for (int j = 0; j < 4; ++j) {
      const __bf16* brow = &Vt[j * 16 + l16][0];
      v8bf blo = *(const v8bf*)(brow + kb + hi * 8);
      v8bf bhi = *(const v8bf*)(brow + kb + 16 + hi * 8);
      v16bf bfr = SHUF16(blo, bhi);
      acc[j] = __builtin_amdgcn_wmma_f32_16x16x32_bf16(
          false, af, false, bfr, (short)0, acc[j], false, false);
    }
  }

  __bf16* dst = (doSin ? Q : P) + ((size_t)bh << 18);   // * 64*64*64
  #pragma unroll
  for (int j = 0; j < 4; ++j) {
    int d = j * 16 + l16;
    #pragma unroll
    for (int r = 0; r < 8; ++r) {
      int vv = ublk * 16 + hi * 8 + r;
      dst[((((size_t)vv << 6) + h) << 6) + d] = (__bf16)acc[j][r];
    }
  }
}

// ---------------------------------------------------------------------------
// FFT stage 2: one block per (bh, v). Y[u,d] = C·P + (-S)·Q, written as
// spectral output O[b, u*64+v, head*64+d] (bf16). Waves: 4 u-blocks x 2 d-halves.
// ---------------------------------------------------------------------------
__global__ __launch_bounds__(256) void fft64_stage2(
    const __bf16* __restrict__ P, const __bf16* __restrict__ Q,
    __bf16* __restrict__ O)
{
  __shared__ __bf16 Cs[64][72];
  __shared__ __bf16 Sn[64][72];   // -sin (NEG bits must be 0 for bf16 WMMA)
  __shared__ __bf16 Pt[64][72];   // [d][h]
  __shared__ __bf16 Qt[64][72];

  const int tid = threadIdx.x;
  const int bid = blockIdx.x;
  const int v  = bid & 63;
  const int bh = bid >> 6;
  const int b = bh / 12, head = bh % 12;

  #pragma unroll
  for (int i = 0; i < 16; ++i) {
    int e = i * 256 + tid;
    int u = e >> 6, hh = e & 63;
    float ang = (float)((u * hh) & 63) * 0.0981747704246810387f;
    float sv, cv;
    __sincosf(ang, &sv, &cv);
    Cs[u][hh] = (__bf16)cv;
    Sn[u][hh] = (__bf16)(-sv);
  }

  {
    const __bf16* psrc = P + ((size_t)(bh * 64 + v) << 12);  // * 64*64
    const __bf16* qsrc = Q + ((size_t)(bh * 64 + v) << 12);
    int hh = tid >> 2;
    int d0 = (tid & 3) << 4;
    size_t off = ((size_t)hh << 6) + d0;
    v8bf p0 = *(const v8bf*)(psrc + off);
    v8bf p1 = *(const v8bf*)(psrc + off + 8);
    v8bf q0 = *(const v8bf*)(qsrc + off);
    v8bf q1 = *(const v8bf*)(qsrc + off + 8);
    #pragma unroll
    for (int i = 0; i < 8; ++i) {
      Pt[d0 + i][hh] = p0[i];  Pt[d0 + 8 + i][hh] = p1[i];
      Qt[d0 + i][hh] = q0[i];  Qt[d0 + 8 + i][hh] = q1[i];
    }
  }
  __syncthreads();

  const int wv = tid >> 5, lane = tid & 31, hi = lane >> 4, l16 = lane & 15;
  const int ublk = wv & 3;
  const int half = wv >> 2;

  v8f zero = {};
  v8f acc[2] = { zero, zero };

  #pragma unroll
  for (int ks = 0; ks < 2; ++ks) {
    int kb = ks * 32;
    const __bf16* crow = &Cs[ublk * 16 + l16][0];
    const __bf16* srow = &Sn[ublk * 16 + l16][0];
    v16bf cf = SHUF16(*(const v8bf*)(crow + kb + hi * 8),
                      *(const v8bf*)(crow + kb + 16 + hi * 8));
    v16bf sf = SHUF16(*(const v8bf*)(srow + kb + hi * 8),
                      *(const v8bf*)(srow + kb + 16 + hi * 8));
    #pragma unroll
    for (int jj = 0; jj < 2; ++jj) {
      int j = half * 2 + jj;
      const __bf16* prow = &Pt[j * 16 + l16][0];
      const __bf16* qrow = &Qt[j * 16 + l16][0];
      v16bf pf = SHUF16(*(const v8bf*)(prow + kb + hi * 8),
                        *(const v8bf*)(prow + kb + 16 + hi * 8));
      v16bf qf = SHUF16(*(const v8bf*)(qrow + kb + hi * 8),
                        *(const v8bf*)(qrow + kb + 16 + hi * 8));
      acc[jj] = __builtin_amdgcn_wmma_f32_16x16x32_bf16(
          false, cf, false, pf, (short)0, acc[jj], false, false);
      acc[jj] = __builtin_amdgcn_wmma_f32_16x16x32_bf16(
          false, sf, false, qf, (short)0, acc[jj], false, false);
    }
  }

  #pragma unroll
  for (int jj = 0; jj < 2; ++jj) {
    int d = (half * 2 + jj) * 16 + l16;
    #pragma unroll
    for (int r = 0; r < 8; ++r) {
      int u = ublk * 16 + hi * 8 + r;
      O[(size_t)(b * 4096 + u * 64 + v) * 768 + head * 64 + d] = (__bf16)acc[jj][r];
    }
  }
}

// ---------------------------------------------------------------------------
extern "C" void kernel_launch(void* const* d_in, const int* in_sizes, int n_in,
                              void* d_out, int out_size, void* d_ws, size_t ws_size,
                              hipStream_t stream)
{
  (void)in_sizes; (void)n_in; (void)out_size;

  const float* x      = (const float*)d_in[0];   // [8,4096,768]
  const float* qkv_w  = (const float*)d_in[1];   // [2304,768]
  const float* proj_w = (const float*)d_in[2];   // [768,768]
  const float* proj_b = (const float*)d_in[3];   // [768]
  float* out = (float*)d_out;                    // [8,4096,768]

  const int Bb = 8, C = 768, N = 4096;
  const int M = Bb * N;                          // 32768
  const size_t seg = (size_t)M * C * sizeof(__bf16);  // 50,331,648 B

  if (ws_size < 3 * seg) return;                 // need ~151 MB of scratch

  __bf16* Vbuf = (__bf16*)d_ws;                         // V16; reused as spectral out
  __bf16* Pbuf = (__bf16*)((char*)d_ws + seg);
  __bf16* Qbuf = (__bf16*)((char*)d_ws + 2 * seg);

  dim3 gGemm(M / 128, C / 64);                   // 256 x 12

  // [1] V = X @ Wv^T   (Wv = qkv_w rows [2C, 3C))
  gemm_f32in_bf16out<<<gGemm, 256, 0, stream>>>(
      x, qkv_w + (size_t)2 * C * C, Vbuf, M, C, C);

  // [2] w-axis DFT:  P = C@V, Q = S@V    (6144 blocks: b*head*h)
  fft64_stage1<<<dim3(Bb * 12 * 64), 256, 0, stream>>>(Vbuf, Pbuf, Qbuf);

  // [3] h-axis DFT:  Y = C@P - S@Q  -> spectral out (reuses Vbuf)
  fft64_stage2<<<dim3(Bb * 12 * 64), 256, 0, stream>>>(Pbuf, Qbuf, Vbuf);

  // [4] out = Y @ proj_w^T + proj_b
  gemm_bf16in_f32out<<<gGemm, 256, 0, stream>>>(
      Vbuf, proj_w, proj_b, out, M, C, C);
}
